// HausdorfLoss_27874337751589
// MI455X (gfx1250) — compile-verified, matrix operand order checked by reading, not run on hardware
//
#include <hip/hip_runtime.h>

typedef unsigned int u32;
typedef float v2f  __attribute__((ext_vector_type(2)));
typedef float v8f  __attribute__((ext_vector_type(8)));
typedef _Float16 v16h __attribute__((ext_vector_type(16)));
typedef u32 v4u __attribute__((ext_vector_type(4)));
typedef u32 v8u __attribute__((ext_vector_type(8)));

#define BATCH 8
#define H 256
#define W 256
#define NPIX (BATCH * H * W)
#define FEPS 1e-8f
#define CAP 100000
#define BIGF 1e10f

// CDNA5 async data-movement toggles (flip to 0 if the assembler rejects them)
#define USE_CDNA5_TDM   1
#define USE_CDNA5_ASYNC 1

// workspace layout (bytes)
#define OFF_PM   0
#define OFF_TM   (OFF_PM + BATCH * H * 8 * 4)
#define OFF_IB   (OFF_TM + BATCH * H * 8 * 4)
#define OFF_LB   (OFF_IB + BATCH * H * 8 * 4)
#define OFF_G    (OFF_LB + BATCH * H * 8 * 4)
#define OFF_PART (OFF_G + BATCH * 2 * H * W * 4)
#define NPART    6144   // 2048 focal-row partials + 4096 rowpass partials (= 96*64)

__device__ __forceinline__ int hd_getbit(const u32* w, int y) {
  return (w[y >> 5] >> (y & 31)) & 1;
}

__device__ __forceinline__ float hd_focal(float xv, float tv) {
  float p = 1.0f / (1.0f + expf(-xv));
  return -(1.0f - p) * tv * logf(p + FEPS)
         - p * (1.0f - tv) * logf(1.0f - p + FEPS);
}

// ---------- kernel 1: pack prediction/target masks into row bitfields ----------
__global__ void hd_mask_kernel(const float* __restrict__ in,
                               const float* __restrict__ tg,
                               u32* __restrict__ pmb, u32* __restrict__ tmb) {
  __shared__ u32 mb[16];
  int bid = blockIdx.x;          // b*H + x
  int y = threadIdx.x;
  if (y < 16) mb[y] = 0u;
  __syncthreads();
  int idx = bid * W + y;
  bool pm = in[idx] > 0.0f;      // sigmoid(x) > 0.5  <=>  x > 0
  bool tm = tg[idx] > 0.5f;
  if (pm) atomicOr(&mb[y >> 5], 1u << (y & 31));
  if (tm) atomicOr(&mb[8 + (y >> 5)], 1u << (y & 31));
  __syncthreads();
  if (y < 8)       pmb[bid * 8 + y] = mb[y];
  else if (y < 16) tmb[bid * 8 + (y - 8)] = mb[y];
}

// ---------- kernel 2: 4-neighbor boundary bits + per-row focal sums ----------
__global__ void hd_boundary_kernel(const float* __restrict__ in,
                                   const float* __restrict__ tg,
                                   const u32* __restrict__ pmb,
                                   const u32* __restrict__ tmb,
                                   u32* __restrict__ ibb, u32* __restrict__ lbb,
                                   float* __restrict__ part) {
  __shared__ u32 nb[2][3][8];    // [mask][row-1..row+1][word]
  __shared__ u32 ob[16];
  __shared__ float red[256];
  int bid = blockIdx.x;          // b*H + x
  int b = bid >> 8, x = bid & 255;
  int t = threadIdx.x;
  if (t < 48) {
    int m = t / 24, r = (t % 24) / 8, w = t % 8;
    int row = x - 1 + r;
    const u32* src = (m == 0) ? pmb : tmb;
    u32 v = 0u;                          // image border counts as background
    if (row >= 0 && row < H) v = src[(b * H + row) * 8 + w];
    nb[m][r][w] = v;
  }
  if (t >= 48 && t < 64) ob[t - 48] = 0u;
  __syncthreads();
  int y = t;
#pragma unroll
  for (int m = 0; m < 2; ++m) {
    const u32* U = nb[m][0];
    const u32* C = nb[m][1];
    const u32* D = nb[m][2];
    int fg = hd_getbit(C, y);
    int up = hd_getbit(U, y);
    int dn = hd_getbit(D, y);
    int lf = (y > 0)     ? hd_getbit(C, y - 1) : 0;
    int rt = (y < W - 1) ? hd_getbit(C, y + 1) : 0;
    int inner = up & dn & lf & rt;
    int bnd = fg & (inner ^ 1);
    if (bnd) atomicOr(&ob[m * 8 + (y >> 5)], 1u << (y & 31));
  }
  int idx = bid * W + y;
  red[t] = hd_focal(in[idx], tg[idx]);
  __syncthreads();
  for (int s = 128; s > 0; s >>= 1) {
    if (t < s) red[t] += red[t + s];
    __syncthreads();
  }
  if (t < 8)       ibb[bid * 8 + t] = ob[t];
  else if (t < 16) lbb[bid * 8 + (t - 8)] = ob[t];
  if (t == 0) part[bid] = red[0];
}

// ---------- kernel 3: per-column nearest-boundary distance (exact 1D EDT) ----------
__global__ void hd_colpass_kernel(const u32* __restrict__ ibb,
                                  const u32* __restrict__ lbb,
                                  int* __restrict__ gi) {
  int b = blockIdx.x >> 1;
  int m = blockIdx.x & 1;        // 0: source = ib, 1: source = lb
  int y = threadIdx.x;
  const u32* sb = ((m == 0) ? ibb : lbb) + b * H * 8;
  int base = ((b * 2 + m) * H) * W + y;
  int d = CAP;
  for (int x = 0; x < H; ++x) {
    u32 wd = sb[x * 8 + (y >> 5)];
    int bit = (wd >> (y & 31)) & 1;
    d = bit ? 0 : ((d < CAP) ? d + 1 : CAP);
    gi[base + x * W] = d;
  }
  d = CAP;
  for (int x = H - 1; x >= 0; --x) {
    u32 wd = sb[x * 8 + (y >> 5)];
    int bit = (wd >> (y & 31)) & 1;
    d = bit ? 0 : ((d < CAP) ? d + 1 : CAP);
    int n = min(d, gi[base + x * W]);
    float fn = (float)n;
    gi[base + x * W] = __float_as_int(fminf(fn * fn, BIGF)); // CAP^2 == 1e10 == BIG
  }
}

// ---------- kernel 4: row min-plus pass + weighted-focal accumulation ----------
// g row staged into LDS by the Tensor Data Mover; consumer bits by async copy.
__global__ void hd_rowpass_kernel(const float* __restrict__ in,
                                  const float* __restrict__ tg,
                                  const u32* __restrict__ ibb,
                                  const u32* __restrict__ lbb,
                                  const float* __restrict__ gf,
                                  float* __restrict__ part) {
  __shared__ float gr[W];
  __shared__ u32 cb[8];
  __shared__ float red[256];
  int bid = blockIdx.x;          // ((b*2+m)*H + x)
  int x = bid & 255;
  int m = (bid >> 8) & 1;
  int b = bid >> 9;
  int y = threadIdx.x;
  const u32* cons = (m == 0) ? lbb : ibb;   // distances-to-ib consumed at lb, and v.v.

#if USE_CDNA5_TDM
  if (threadIdx.x < 32) {        // one wave issues the TDM descriptor
    unsigned long long ga = (unsigned long long)(size_t)(gf + bid * W);
    u32 ga_lo = (u32)ga;
    u32 ga_hi = (u32)(ga >> 32) & 0x01FFFFFFu;     // addr bits 56:32
    // D# group 0: count=1 | lds_addr | global_addr | type=2 (bits 127:126)
    v4u g0 = { 1u, (u32)(size_t)&gr[0], ga_lo, ga_hi | 0x80000000u };
    // D# group 1: 256x1 f32 tensor == tile; data_size=4B (code 2)
    v8u g1 = { (2u << 16),            // wg_mask=0, data_size=2 (4 bytes)
               ((u32)W) << 16,        // tensor_dim0[15:0] = 256   (bits 79:48)
               (1u << 16),            // tensor_dim0[31:16]=0 | tensor_dim1[15:0]=1
               ((u32)W) << 16,        // tensor_dim1[31:16]=0 | tile_dim0 = 256
               1u,                    // tile_dim1 = 1, tile_dim2 = 0
               (u32)W,                // tensor_dim0_stride[31:0] = 256
               ((u32)W) << 16,        // stride0[47:32]=0 | tensor_dim1_stride[15:0]=256
               0u };                  // tensor_dim1_stride[47:16] = 0
    asm volatile("tensor_load_to_lds %0, %1" :: "s"(g0), "s"(g1) : "memory");
  }
#else
  gr[y] = gf[bid * W + y];
#endif

#if USE_CDNA5_ASYNC
  if (y < 8) {                   // 8 words of consumer-boundary bits
    u32 ldsoff = (u32)(size_t)&cb[y];
    const u32* gptr = cons + (b * H + x) * 8 + y;
    asm volatile("global_load_async_to_lds_b32 %0, %1, off"
                 :: "v"(ldsoff), "v"(gptr) : "memory");
  }
#else
  if (y < 8) cb[y] = cons[(b * H + x) * 8 + y];
#endif

#if USE_CDNA5_TDM
  __builtin_amdgcn_s_wait_tensorcnt(0);
#endif
#if USE_CDNA5_ASYNC
  asm volatile("s_wait_asynccnt 0x0" ::: "memory");
#endif
  __syncthreads();

  float fy = (float)y;
  float d2 = 1e30f;
  float fyp = 0.0f;
#pragma unroll 4
  for (int yp = 0; yp < W; ++yp) {
    float diff = fy - fyp;
    d2 = fminf(d2, fmaf(diff, diff, gr[yp]));
    fyp += 1.0f;
  }
  int bit = (cb[y >> 5] >> (y & 31)) & 1;
  int idx = (b * H + x) * W + y;
  float fl = hd_focal(in[idx], tg[idx]);
  float val = bit ? expf(-sqrtf(d2)) * fl : 0.0f;   // THETA = SIGMA = 1
  red[y] = val;
  __syncthreads();
  for (int s = 128; s > 0; s >>= 1) {
    if (y < s) red[y] += red[y + s];
    __syncthreads();
  }
  if (y == 0) part[2048 + bid] = red[0];
}

// ---------- kernel 5: exact WMMA f32 reduction of all partials ----------
__global__ void hd_final_wmma_kernel(const float* __restrict__ part,
                                     float* __restrict__ out) {
  __shared__ float r32[32];
  int lane = threadIdx.x;        // single wave32, EXEC all ones (WMMA requirement)
  v8f acc = {0.f, 0.f, 0.f, 0.f, 0.f, 0.f, 0.f, 0.f};
#if __has_builtin(__builtin_amdgcn_wmma_f32_16x16x4_f32)
  v2f bm = {1.0f, 1.0f};         // all-ones B => D rows accumulate row-sums of A
  for (int base = 0; base < NPART; base += 64) {
    v2f a;
    a[0] = part[base + 2 * lane];
    a[1] = part[base + 2 * lane + 1];
    acc = __builtin_amdgcn_wmma_f32_16x16x4_f32(false, a, false, bm,
                                                (short)0, acc, false, false);
  }
#else
  v16h bm16;
#pragma unroll
  for (int i = 0; i < 16; ++i) bm16[i] = (_Float16)1.0f;
  for (int base = 0; base < NPART; base += 512) {
    v16h a;
#pragma unroll
    for (int i = 0; i < 16; ++i) a[i] = (_Float16)part[base + 16 * lane + i];
    acc = __builtin_amdgcn_wmma_f32_16x16x32_f16(false, a, false, bm16,
                                                 (short)0, acc, false, false);
  }
#endif
  float s = 0.0f;
#pragma unroll
  for (int c = 0; c < 8; ++c) s += acc[c];
  r32[lane] = s;
  __syncthreads();
  if (lane == 0) {
    float tot = 0.0f;
    for (int i = 0; i < 32; ++i) tot += r32[i];
    // every D column replicates the total -> divide by 16, then take the mean
    out[0] = tot * (1.0f / 16.0f) / (float)NPIX;
  }
}

extern "C" void kernel_launch(void* const* d_in, const int* in_sizes, int n_in,
                              void* d_out, int out_size, void* d_ws, size_t ws_size,
                              hipStream_t stream) {
  const float* in = (const float*)d_in[0];
  const float* tg = (const float*)d_in[1];
  char* ws = (char*)d_ws;
  u32* pmb  = (u32*)(ws + OFF_PM);
  u32* tmb  = (u32*)(ws + OFF_TM);
  u32* ibb  = (u32*)(ws + OFF_IB);
  u32* lbb  = (u32*)(ws + OFF_LB);
  int* gi   = (int*)(ws + OFF_G);
  float* gf = (float*)(ws + OFF_G);
  float* part = (float*)(ws + OFF_PART);
  float* out = (float*)d_out;

  hd_mask_kernel<<<BATCH * H, 256, 0, stream>>>(in, tg, pmb, tmb);
  hd_boundary_kernel<<<BATCH * H, 256, 0, stream>>>(in, tg, pmb, tmb, ibb, lbb, part);
  hd_colpass_kernel<<<BATCH * 2, 256, 0, stream>>>(ibb, lbb, gi);
  hd_rowpass_kernel<<<BATCH * 2 * H, 256, 0, stream>>>(in, tg, ibb, lbb, gf, part);
  hd_final_wmma_kernel<<<1, 32, 0, stream>>>(part, out);
}
